// BezierParameterProcessor_32186484916309
// MI455X (gfx1250) — matrix operand order repaired and checked by reading, not compiled, not run on hardware
//
#include <hip/hip_runtime.h>
#include <hip/hip_bf16.h>

typedef __attribute__((ext_vector_type(16))) _Float16 v16h;
typedef __attribute__((ext_vector_type(8)))  float    v8f;

#define H_RES   256
#define W_RES   256
#define HW      (H_RES * W_RES)
#define NPTS    128          // 16 curves * 8 control points
#define NBATCH  16
#define LOG2E   1.44269504088896340736f
#define SQRT2PI 2.50662827463100050242f

// Raw v_exp_f32: args here are <= ~1e-7, and anything below 2^-126 is flushed
// by the f16 A-fragment conversion anyway, so skipping the libm denorm-safe
// expansion (v_cmp/cndmask/ldexp scaffolding) is numerically harmless.
#if __has_builtin(__builtin_amdgcn_exp2f)
#define FAST_EXP2(x) __builtin_amdgcn_exp2f(x)
#else
#define FAST_EXP2(x) exp2f(x)
#endif

// ---------------------------------------------------------------------------
// Kernel 1: point-encoder MLP  (2 -> 64 -> 128 -> 128 -> 2), one thread/point.
// 2048 points total => ~0.1% of total FLOPs; plain VALU is the right tool.
// ---------------------------------------------------------------------------
__global__ __launch_bounds__(64) void bezier_mlp_kernel(
    const float* __restrict__ pts,                      // [B*N, 2]
    const float* __restrict__ W1, const float* __restrict__ b1,
    const float* __restrict__ W2, const float* __restrict__ b2,
    const float* __restrict__ W3, const float* __restrict__ b3,
    const float* __restrict__ Wf, const float* __restrict__ bf,
    float* __restrict__ vecs)                           // [B*N, 2]
{
    __shared__ float sh2[128 * 64];                     // h2, column per thread
    const int tid = threadIdx.x;
    const int p   = blockIdx.x * 64 + tid;              // global point id

    const float x = pts[p * 2 + 0];
    const float y = pts[p * 2 + 1];

    // layer 1: 2 -> 64, ReLU (h1 kept in VGPRs)
    float h1[64];
#pragma unroll
    for (int j = 0; j < 64; ++j)
        h1[j] = fmaxf(fmaf(W1[j * 2 + 0], x, fmaf(W1[j * 2 + 1], y, b1[j])), 0.0f);

    // layer 2: 64 -> 128, ReLU (h2 in LDS; each thread owns its column)
    for (int j = 0; j < 128; ++j) {
        float a = b2[j];
#pragma unroll
        for (int i = 0; i < 64; ++i)
            a = fmaf(W2[j * 64 + i], h1[i], a);
        sh2[j * 64 + tid] = fmaxf(a, 0.0f);
    }
    // no barrier needed: each thread reads only its own column

    // layer 3 (no ReLU) fused with final 128 -> 2 projection
    float vx = bf[0], vy = bf[1];
    for (int j = 0; j < 128; ++j) {
        float e = b3[j];
        for (int i = 0; i < 128; ++i)
            e = fmaf(W3[j * 128 + i], sh2[i * 64 + tid], e);
        vx = fmaf(Wf[j],       e, vx);
        vy = fmaf(Wf[128 + j], e, vy);
    }
    vecs[p * 2 + 0] = vx;
    vecs[p * 2 + 1] = vy;
}

// ---------------------------------------------------------------------------
// Kernel 2: KDE density + control field via WMMA.
//
// Per 16-pixel grid tile the three reductions over N=128 points
//     density[g] = sum_n e_n,  field[g,c] = sum_n e_n * v_nc
// are [16 x 32] x [32 x 16] f16 GEMMs (v_wmma_f32_16x16x32_f16, f32 acc):
//     B col0 = exp(1e-8/2bw^2)  (density/field exponent ratio)
//     B col1 = vx_n * valid_n,  B col2 = vy_n * valid_n,  cols 3..15 = 0.
//
// Gaussian exponent in expanded-dot form with per-point precompute:
//     arg = cx_k*gx + cy_k*gy + cc_k + c*|g|^2,
//     cx = -2c*px, cy = -2c*py, cc = c*(|p|^2 + 1e-8),  c = -log2e/(2bw^2)
// => 1 ds_load_b128 + 3 FMA-class VALU + 1 raw v_exp_f32 + 1/2 cvt_pk per
//    gaussian. Purely compute-bound (12 MB of output vs 23.3 TB/s HBM).
// ---------------------------------------------------------------------------
__global__ __launch_bounds__(256) void kde_field_wmma_kernel(
    const float* __restrict__ pts,     // [B*N, 2]
    const float* __restrict__ vecs,    // [B*N, 2] from MLP
    const float* __restrict__ bw_p,    // scalar
    const float* __restrict__ thr_p,   // scalar
    float* __restrict__ out)           // density [B*HW] ++ field [B*2*HW]
{
    __shared__ float4 s_kc[NPTS];              // (cx, cy, cc, 0) per point
    __shared__ float  s_vx[NPTS], s_vy[NPTS];  // masked field vectors

    const int b   = blockIdx.y;
    const int tid = threadIdx.x;

    const float bw      = fmaxf(bw_p[0], 1e-5f);
    const float inv2bw2 = 1.0f / (2.0f * bw * bw);
    const float cexp    = -LOG2E * inv2bw2;               // exp(x)=exp2(x*log2e)
    const float ratio   = FAST_EXP2(1e-8f * inv2bw2 * LOG2E);
    const float dnorm   = 1.0f / ((float)NPTS * bw * SQRT2PI);
    const float thr     = thr_p[0];

    for (int i = tid; i < NPTS; i += 256) {
        const float px = pts[(b * NPTS + i) * 2 + 0];
        const float py = pts[(b * NPTS + i) * 2 + 1];
        const float cc = cexp * fmaf(px, px, fmaf(py, py, 1e-8f));
        s_kc[i] = make_float4(-2.0f * cexp * px, -2.0f * cexp * py, cc, 0.0f);
        const float valid = (fabsf(px) > 1e-8f || fabsf(py) > 1e-8f) ? 1.0f : 0.0f;
        s_vx[i] = vecs[(b * NPTS + i) * 2 + 0] * valid;
        s_vy[i] = vecs[(b * NPTS + i) * 2 + 1] * valid;
    }
    __syncthreads();

    const int lane    = tid & 31;
    const int wave    = tid >> 5;       // 0..7
    const int halfsel = lane >> 4;      // half-wave select
    const int col     = lane & 15;      // B/C/D column (and A row) for this lane

    // Batch-constant B fragments, one v16h per 32-wide K chunk.
    // B k order used here (32x16, contiguous): lanes 0-15 hold K=0..15 of column `lane`,
    // lanes 16-31 hold K=16..31 of column `lane-16`.
    v16h bfrag[4];
#pragma unroll
    for (int s = 0; s < 4; ++s) {
        v16h f;
#pragma unroll
        for (int e = 0; e < 16; ++e) {
            const int k = s * 32 + halfsel * 16 + e;
            float v = 0.0f;
            if (col == 0)      v = ratio;
            else if (col == 1) v = s_vx[k];
            else if (col == 2) v = s_vy[k];
            f[e] = (_Float16)v;
        }
        bfrag[s] = f;
    }

    float* __restrict__ outD  = out + (size_t)b * HW;
    float* __restrict__ outFx = out + (size_t)NBATCH * HW + (size_t)b * 2 * HW;
    float* __restrict__ outFy = outFx + HW;

    const int wave_g = blockIdx.x * 8 + wave;
    const int nwaves = gridDim.x * 8;

    for (int tile = wave_g; tile < HW / 16; tile += nwaves) {
        const int g0 = tile * 16;
        const int g  = g0 + col;        // this lane's grid row (A: M = lane&15)
        const float gx = -1.0f + (float)(g & (W_RES - 1)) * (2.0f / (W_RES - 1));
        const float gy = -1.0f + (float)(g >> 8)          * (2.0f / (H_RES - 1));
        const float base = cexp * fmaf(gx, gx, gy * gy);   // c*|g|^2

        v8f acc = {};
#pragma unroll
        for (int s = 0; s < 4; ++s) {
            // A k order used here (16x32, contiguous): lanes 0-15: elems 0-15 -> K 0-15,
            // lanes 16-31: elems 0-15 -> K 16-31 (matches B).
            v16h a;
#pragma unroll
            for (int e = 0; e < 16; ++e) {
                const int k = s * 32 + halfsel * 16 + e;
                const float4 kc = s_kc[k];
                const float arg = fmaf(kc.x, gx, fmaf(kc.y, gy, kc.z + base));
                const float ev  = FAST_EXP2(arg);   // exp(-(d2+1e-8)/(2bw^2))
                a[e] = (_Float16)ev;
            }
            acc = __builtin_amdgcn_wmma_f32_16x16x32_f16(
                /*neg_a=*/false, a, /*neg_b=*/false, bfrag[s],
                /*c_mod=*/(short)0, acc, /*reuse_a=*/false, /*reuse_b=*/false);
        }

        // D layout: lane = column, VGPR r = row r (+8 for upper half-wave).
        if (col < 3) {
#pragma unroll
            for (int r = 0; r < 8; ++r) {
                const int gg = g0 + halfsel * 8 + r;
                const float v = acc[r];
                if (col == 0) {
                    const float d = fmaf(v, dnorm, -thr);
                    outD[gg] = 1.0f / (1.0f + FAST_EXP2(-d * LOG2E));  // sigmoid
                } else if (col == 1) {
                    outFx[gg] = tanhf(v);
                } else {
                    outFy[gg] = tanhf(v);
                }
            }
        }
    }
}

// ---------------------------------------------------------------------------
// Launch: inputs in setup_inputs() order:
// 0 bezier_points, 1 W1, 2 b1, 3 W2, 4 b2, 5 W3, 6 b3, 7 Wf, 8 bf,
// 9 kde_bandwidth, 10 density_threshold.
// Output: density_map [B,1,H,W] ++ field_map [B,2,H,W], flat f32.
// ---------------------------------------------------------------------------
extern "C" void kernel_launch(void* const* d_in, const int* in_sizes, int n_in,
                              void* d_out, int out_size, void* d_ws, size_t ws_size,
                              hipStream_t stream) {
    const float* pts = (const float*)d_in[0];
    const float* W1  = (const float*)d_in[1];
    const float* b1  = (const float*)d_in[2];
    const float* W2  = (const float*)d_in[3];
    const float* b2  = (const float*)d_in[4];
    const float* W3  = (const float*)d_in[5];
    const float* b3  = (const float*)d_in[6];
    const float* Wf  = (const float*)d_in[7];
    const float* bf  = (const float*)d_in[8];
    const float* bw  = (const float*)d_in[9];
    const float* thr = (const float*)d_in[10];

    float* vecs = (float*)d_ws;   // [B*N, 2] = 16 KB scratch

    // Phase 1: MLP over all 2048 points (32 blocks x 64 threads)
    bezier_mlp_kernel<<<dim3((NBATCH * NPTS) / 64), 64, 0, stream>>>(
        pts, W1, b1, W2, b2, W3, b3, Wf, bf, vecs);

    // Phase 2: KDE + field. grid.y = batch; 64 blocks x 8 waves = 512 waves
    // per batch -> 8 sixteen-pixel tiles per wave (HW/16 = 4096 tiles).
    kde_field_wmma_kernel<<<dim3(64, NBATCH), 256, 0, stream>>>(
        pts, vecs, bw, thr, (float*)d_out);
}
